// AvaAttention_30279519436976
// MI455X (gfx1250) — matrix operand order repaired and checked by reading, not compiled, not run on hardware
//
#include <hip/hip_runtime.h>

// ---------------- types ----------------
typedef __bf16 bf16_t;
typedef __attribute__((ext_vector_type(16))) __bf16 v16bf;
typedef __attribute__((ext_vector_type(2)))  __bf16 bf16x2;
typedef __attribute__((ext_vector_type(8)))  float  v8f;
typedef __attribute__((ext_vector_type(2)))  float  f32x2;

#define DEVINL __device__ __forceinline__

// Problem constants
enum { B_ = 2, T_ = 2048, HID_ = 2048, NH_ = 16, KVH_ = 4, HD_ = 128 };

// ---------------- bf16 helpers (native cvt) ----------------
DEVINL bf16_t f2bf16(float f) { return (bf16_t)f; }

// single v_cvt_pk_bf16_f32
DEVINL unsigned pack2(float a, float b) {
  f32x2 fv; fv.x = a; fv.y = b;
  bf16x2 h = __builtin_convertvector(fv, bf16x2);
  return __builtin_bit_cast(unsigned, h);
}

// ---------------- CDNA5 async copy: global -> LDS, 16B per lane ----------------
DEVINL void async_copy_b128(const bf16_t* gsrc, bf16_t* ldst) {
  unsigned lds = (unsigned)(uintptr_t)ldst;            // low 32 bits = LDS offset
  unsigned long long ga = (unsigned long long)(uintptr_t)gsrc;
  asm volatile("global_load_async_to_lds_b128 %0, %1, off"
               :: "v"(lds), "v"(ga) : "memory");
}
DEVINL void wait_async0() {
  asm volatile("s_wait_asynccnt 0x0" ::: "memory");
}

// ---------------- WMMA fragments ----------------
struct Frag { union { v16bf v; unsigned u[8]; }; };

// A-operand (16xK slice, 16-bit): row r0+(lane&15); VGPR0-3: K = k0+8*half+2i,
// VGPR4-7: K = k0+16+8*half+2(i-4).  base row-major [rows][ldk] bf16, ldk even.
DEVINL void load_frag_a(Frag& f, const bf16_t* base, int r0, int ldk, int k0, int lane) {
  const int half = lane >> 4;
  const int m = r0 + (lane & 15);
  const unsigned* p = (const unsigned*)(base + m * ldk);
  const int kb = (k0 >> 1) + 4 * half;
  f.u[0] = p[kb + 0]; f.u[1] = p[kb + 1]; f.u[2] = p[kb + 2]; f.u[3] = p[kb + 3];
  f.u[4] = p[kb + 8]; f.u[5] = p[kb + 9]; f.u[6] = p[kb + 10]; f.u[7] = p[kb + 11];
}

// B-operand (Kx16 slice, 16-bit): column n0+(lane&15); VGPR0-3: K = k0+8*half+2i,
// VGPR4-7: K = k0+16+8*half+2(i-4).  base is [n][ldk] bf16 (K-contiguous per column).
DEVINL void load_frag_b(Frag& f, const bf16_t* base, int n0, int ldk, int k0, int lane) {
  const int half = lane >> 4;
  const int n = n0 + (lane & 15);
  const unsigned* p = (const unsigned*)(base + n * ldk);
  const int kb = (k0 >> 1) + 4 * half;
#pragma unroll
  for (int i = 0; i < 8; ++i) f.u[i] = p[kb + i + (i >= 4 ? 4 : 0)];
}

DEVINL v8f wmma_bf16(const Frag& a, const Frag& b, v8f c) {
  return __builtin_amdgcn_wmma_f32_16x16x32_bf16(false, a.v, false, b.v,
                                                 (short)0, c, false, false);
}

// =====================================================================
// Kernel 1: QKV projection + RoPE.  Block tile 64x128, BK=64, 8 waves.
// grid.x = (B*T)/64 = 64, grid.y = 24 (16 Q-heads | 4 K-heads | 4 V-heads)
// V is written TRANSPOSED: v_ws[b][kh][d][t]  (HD x T per head)
// =====================================================================
__global__ __launch_bounds__(256) void k_qkv_rope(
    const float* __restrict__ X,
    const float* __restrict__ Wq, const float* __restrict__ Wk,
    const float* __restrict__ Wv,
    const float* __restrict__ cosp, const float* __restrict__ sinp,
    bf16_t* __restrict__ q_ws, bf16_t* __restrict__ k_ws,
    bf16_t* __restrict__ v_ws) {
  __shared__ __align__(16) bf16_t As[64][64];    // 8 KB
  __shared__ __align__(16) bf16_t Bs[128][64];   // 16 KB (K-contiguous per col)
  __shared__ float Cs[64][128];                  // 32 KB result tile

  const int tid = threadIdx.x;
  const int mb = blockIdx.x;
  const int cb = blockIdx.y;

  const float* W; int ldw, col0, region, head;
  if (cb < 16)      { W = Wq; ldw = 2048; col0 = cb * 128;        region = 0; head = cb; }
  else if (cb < 20) { W = Wk; ldw = 512;  col0 = (cb - 16) * 128; region = 1; head = cb - 16; }
  else              { W = Wv; ldw = 512;  col0 = (cb - 20) * 128; region = 2; head = cb - 20; }

  const int wave = tid >> 5, lane = tid & 31;
  const int wm = (wave >> 2) * 32, wn = (wave & 3) * 32;
  const int hf = lane >> 4, ln = lane & 15;

  v8f acc[2][2] = {};

  for (int it = 0; it < HID_ / 64; ++it) {
    // ---- phase 1: issue all global loads (keeps many in flight) ----
    float4 xa[4], wb[8];
#pragma unroll
    for (int r = 0; r < 4; ++r) {
      int q = tid + 256 * r;                 // 0..1023
      int m = q >> 4, k4 = (q & 15) << 2;
      xa[r] = *(const float4*)(X + (size_t)(mb * 64 + m) * HID_ + it * 64 + k4);
    }
#pragma unroll
    for (int r = 0; r < 8; ++r) {
      int q = tid + 256 * r;                 // 0..2047
      int k = q >> 5, n4 = (q & 31) << 2;
      wb[r] = *(const float4*)(W + (size_t)(it * 64 + k) * ldw + col0 + n4);
    }
    // prefetch next weight slab (global_prefetch_b8)
    if (it + 1 < HID_ / 64)
      __builtin_prefetch(W + (size_t)((it + 1) * 64 + (tid >> 2)) * ldw + col0 +
                             (tid & 3) * 32, 0, 0);

    // ---- phase 2: convert + LDS stores ----
#pragma unroll
    for (int r = 0; r < 4; ++r) {
      int q = tid + 256 * r;
      int m = q >> 4, k4 = (q & 15) << 2;
      uint2 pk; pk.x = pack2(xa[r].x, xa[r].y); pk.y = pack2(xa[r].z, xa[r].w);
      *(uint2*)(&As[m][k4]) = pk;
    }
#pragma unroll
    for (int r = 0; r < 8; ++r) {
      int q = tid + 256 * r;
      int k = q >> 5, n4 = (q & 31) << 2;
      Bs[n4 + 0][k] = f2bf16(wb[r].x);
      Bs[n4 + 1][k] = f2bf16(wb[r].y);
      Bs[n4 + 2][k] = f2bf16(wb[r].z);
      Bs[n4 + 3][k] = f2bf16(wb[r].w);
    }
    __syncthreads();

    Frag a0, a1, b0, b1;
#pragma unroll
    for (int k0 = 0; k0 < 64; k0 += 32) {
      load_frag_a(a0, &As[0][0], wm,      64, k0, lane);
      load_frag_a(a1, &As[0][0], wm + 16, 64, k0, lane);
      load_frag_b(b0, &Bs[0][0], wn,      64, k0, lane);
      load_frag_b(b1, &Bs[0][0], wn + 16, 64, k0, lane);
      acc[0][0] = wmma_bf16(a0, b0, acc[0][0]);
      acc[0][1] = wmma_bf16(a0, b1, acc[0][1]);
      acc[1][0] = wmma_bf16(a1, b0, acc[1][0]);
      acc[1][1] = wmma_bf16(a1, b1, acc[1][1]);
    }
    __syncthreads();
  }

  // dump accumulators to LDS f32 tile
#pragma unroll
  for (int i = 0; i < 2; ++i)
#pragma unroll
    for (int j = 0; j < 2; ++j)
#pragma unroll
      for (int r = 0; r < 8; ++r)
        Cs[wm + 16 * i + r + 8 * hf][wn + 16 * j + ln] = acc[i][j][r];
  __syncthreads();

  // epilogue: RoPE (Q/K) or passthrough (V, transposed), scatter bf16
#pragma unroll
  for (int e = tid; e < 64 * 128; e += 256) {
    int m = e >> 7, d = e & 127;
    int gr = mb * 64 + m;
    int b = gr >> 11, t = gr & (T_ - 1);
    if (region == 2) {
      // V transposed: [b][kh][d][t]
      v_ws[(((size_t)b * KVH_ + head) * HD_ + d) * T_ + t] = f2bf16(Cs[m][d]);
    } else {
      float y = Cs[m][d];
      float part = (d < 64) ? -Cs[m][d + 64] : Cs[m][d - 64];
      float val = y * cosp[t * HD_ + d] + part * sinp[t * HD_ + d];
      bf16_t o = f2bf16(val);
      if (region == 0)
        q_ws[(((size_t)b * NH_ + head) * T_ + t) * HD_ + d] = o;
      else
        k_ws[(((size_t)b * KVH_ + head) * T_ + t) * HD_ + d] = o;
    }
  }
}

// =====================================================================
// Kernel 2: flash attention, Br=Bc=64, causal.  grid = (T/64, NH, B)
// wave w: rows 16*(w>>1); S-cols 32*(w&1)+{0,16}; O-cols 64*(w&1)+{0..48}
// Q/K/V tiles staged via CDNA5 async global->LDS copies (ASYNCcnt);
// Q fragments preloaded once (loop-invariant).
// =====================================================================
__global__ __launch_bounds__(256) void k_attn(
    const bf16_t* __restrict__ q_ws, const bf16_t* __restrict__ k_ws,
    const bf16_t* __restrict__ v_ws, bf16_t* __restrict__ ctx_ws) {
  __shared__ __align__(16) bf16_t Qs[64][128];   // 16 KB
  __shared__ __align__(16) bf16_t Ks[64][128];   // 16 KB
  __shared__ __align__(16) bf16_t Vt[128][64];   // 16 KB (V^T tile [d][kv])
  __shared__ __align__(16) float  Ss[64][64];    // 16 KB
  __shared__ __align__(16) bf16_t Ps[64][64];    // 8 KB
  __shared__ float  rowscale[64];
  __shared__ float  rowl[64];

  const int tid = threadIdx.x, lane = tid & 31, wave = tid >> 5;
  const int qb = blockIdx.x, h = blockIdx.y, b = blockIdx.z;
  const int kh = h >> 2;  // NH/KVH = 4
  const int rowbase = 16 * (wave >> 1);
  const int hf = lane >> 4, ln = lane & 15;
  const float scale = 0.08838834764831845f;  // 1/sqrt(128)

  // async-stage Q tile (contiguous 16 KB): 4 x b128 per thread
  const bf16_t* qg = q_ws + (((size_t)b * NH_ + h) * T_ + qb * 64) * HD_;
#pragma unroll
  for (int r = 0; r < 4; ++r)
    async_copy_b128(qg + tid * 8 + r * 2048, &Qs[0][0] + tid * 8 + r * 2048);
  wait_async0();
  __syncthreads();

  // preload loop-invariant Q fragments (32 VGPRs)
  Frag qa[4];
#pragma unroll
  for (int ks = 0; ks < 4; ++ks)
    load_frag_a(qa[ks], &Qs[0][0], rowbase, 128, 32 * ks, lane);

  v8f O[4] = {};
  float m_i = -1e30f, l_i = 0.f;

  for (int jb = 0; jb <= qb; ++jb) {
    // K tile: contiguous rows -> 4 async b128 per thread
    const bf16_t* kg = k_ws + (((size_t)b * KVH_ + kh) * T_ + jb * 64) * HD_;
#pragma unroll
    for (int r = 0; r < 4; ++r)
      async_copy_b128(kg + tid * 8 + r * 2048, &Ks[0][0] + tid * 8 + r * 2048);
    // V^T tile: 128 rows x first 32 contiguous bf16 (kv 0..31) -> 2 async b128 per thread
    const bf16_t* vg = v_ws + ((size_t)b * KVH_ + kh) * HD_ * T_ + jb * 64;
#pragma unroll
    for (int r = 0; r < 2; ++r) {
      int c = tid + 256 * r;                 // 0..511 chunks of 8 bf16
      int d = c >> 2, cc = (c & 3) * 8;
      async_copy_b128(vg + (size_t)d * T_ + cc, &Vt[d][cc]);
    }
    wait_async0();
    __syncthreads();

    // ---- S = Q K^T (scaled, causal-masked) ----
#pragma unroll
    for (int jj = 0; jj < 2; ++jj) {
      int n0 = 32 * (wave & 1) + 16 * jj;
      v8f s = {};
#pragma unroll
      for (int ks = 0; ks < 4; ++ks) {
        Frag kb;
        load_frag_b(kb, &Ks[0][0], n0, 128, 32 * ks, lane);
        s = wmma_bf16(qa[ks], kb, s);
      }
#pragma unroll
      for (int r = 0; r < 8; ++r) {
        int row = rowbase + r + 8 * hf;
        int col = n0 + ln;
        float v = s[r] * scale;
        if (jb * 64 + col > qb * 64 + row) v = -1e30f;  // causal
        Ss[row][col] = v;
      }
    }
    __syncthreads();

    // ---- online softmax (one thread per row, vectorized) ----
    if (tid < 64) {
      const float4* srow = (const float4*)(&Ss[tid][0]);
      unsigned* prow = (unsigned*)(&Ps[tid][0]);
      float mx = m_i;
#pragma unroll
      for (int c4 = 0; c4 < 16; ++c4) {
        float4 s4 = srow[c4];
        mx = fmaxf(mx, fmaxf(fmaxf(s4.x, s4.y), fmaxf(s4.z, s4.w)));
      }
      float corr = __expf(m_i - mx);
      float sum = 0.f;
#pragma unroll
      for (int c4 = 0; c4 < 16; ++c4) {
        float4 s4 = srow[c4];
        float p0 = __expf(s4.x - mx), p1 = __expf(s4.y - mx);
        float p2 = __expf(s4.z - mx), p3 = __expf(s4.w - mx);
        sum += (p0 + p1) + (p2 + p3);
        prow[c4 * 2 + 0] = pack2(p0, p1);
        prow[c4 * 2 + 1] = pack2(p2, p3);
      }
      l_i = l_i * corr + sum;
      m_i = mx;
      rowscale[tid] = corr;
    }
    __syncthreads();

    // ---- rescale running O ----
    float sc[8];
#pragma unroll
    for (int r = 0; r < 8; ++r) sc[r] = rowscale[rowbase + r + 8 * hf];
#pragma unroll
    for (int jj = 0; jj < 4; ++jj)
#pragma unroll
      for (int r = 0; r < 8; ++r) O[jj][r] *= sc[r];

    // ---- O += P V ----
    Frag pa[2];
    load_frag_a(pa[0], &Ps[0][0], rowbase, 64, 0, lane);
    load_frag_a(pa[1], &Ps[0][0], rowbase, 64, 32, lane);
#pragma unroll
    for (int jj = 0; jj < 4; ++jj) {
      int n0 = 64 * (wave & 1) + 16 * jj;
#pragma unroll
      for (int ks = 0; ks < 2; ++ks) {
        Frag vb;
        load_frag_b(vb, &Vt[0][0], n0, 64, 32 * ks, lane);
        O[jj] = wmma_bf16(pa[ks], vb, O[jj]);
      }
    }
    __syncthreads();
  }

  if (tid < 64) rowl[tid] = l_i;
  __syncthreads();

  // normalize, write ctx as bf16 in [b][t][h][d] (matches transpose+reshape)
#pragma unroll
  for (int jj = 0; jj < 4; ++jj) {
    int n0 = 64 * (wave & 1) + 16 * jj;
#pragma unroll
    for (int r = 0; r < 8; ++r) {
      int row = rowbase + r + 8 * hf;
      float val = O[jj][r] / rowl[row];
      ctx_ws[(((size_t)b * T_ + qb * 64 + row) * NH_ + h) * HD_ + n0 + ln] =
          f2bf16(val);
    }
  }
}

// =====================================================================
// Kernel 3: output projection: ctx(bf16, 4096x2048) @ Wo(f32) -> out(f32)
// grid = (64, 16); tile 64x128, BK=64; A staged via async global->LDS.
// =====================================================================
__global__ __launch_bounds__(256) void k_outproj(
    const bf16_t* __restrict__ ctx, const float* __restrict__ Wo,
    float* __restrict__ out) {
  __shared__ __align__(16) bf16_t As[64][64];    // 8 KB
  __shared__ __align__(16) bf16_t Bs[128][64];   // 16 KB

  const int tid = threadIdx.x;
  const int mb = blockIdx.x, nb = blockIdx.y;
  const int wave = tid >> 5, lane = tid & 31;
  const int wm = (wave >> 2) * 32, wn = (wave & 3) * 32;
  const int hf = lane >> 4, ln = lane & 15;

  v8f acc[2][2] = {};

  for (int it = 0; it < HID_ / 64; ++it) {
    // A already bf16: async copy 64 rows x 128B (row stride 4 KB in global)
#pragma unroll
    for (int r = 0; r < 2; ++r) {
      int q = tid + 256 * r;                 // 0..511
      int m = q >> 3, c = (q & 7) * 8;       // 8 bf16 per chunk
      async_copy_b128(ctx + (size_t)(mb * 64 + m) * HID_ + it * 64 + c,
                      &As[m][c]);
    }
    // phase 1: batch-load Wo slab
    float4 wb[8];
#pragma unroll
    for (int r = 0; r < 8; ++r) {
      int q = tid + 256 * r;                 // 0..2047
      int k = q >> 5, n4 = (q & 31) << 2;
      wb[r] = *(const float4*)(Wo + (size_t)(it * 64 + k) * HID_ + nb * 128 + n4);
    }
    if (it + 1 < HID_ / 64)
      __builtin_prefetch(Wo + (size_t)((it + 1) * 64 + (tid >> 2)) * HID_ +
                             nb * 128 + (tid & 3) * 32, 0, 0);
    // phase 2: convert + transposed LDS stores
#pragma unroll
    for (int r = 0; r < 8; ++r) {
      int q = tid + 256 * r;
      int k = q >> 5, n4 = (q & 31) << 2;
      Bs[n4 + 0][k] = f2bf16(wb[r].x);
      Bs[n4 + 1][k] = f2bf16(wb[r].y);
      Bs[n4 + 2][k] = f2bf16(wb[r].z);
      Bs[n4 + 3][k] = f2bf16(wb[r].w);
    }
    wait_async0();
    __syncthreads();

    Frag a0, a1, b0, b1;
#pragma unroll
    for (int k0 = 0; k0 < 64; k0 += 32) {
      load_frag_a(a0, &As[0][0], wm,      64, k0, lane);
      load_frag_a(a1, &As[0][0], wm + 16, 64, k0, lane);
      load_frag_b(b0, &Bs[0][0], wn,      64, k0, lane);
      load_frag_b(b1, &Bs[0][0], wn + 16, 64, k0, lane);
      acc[0][0] = wmma_bf16(a0, b0, acc[0][0]);
      acc[0][1] = wmma_bf16(a0, b1, acc[0][1]);
      acc[1][0] = wmma_bf16(a1, b0, acc[1][0]);
      acc[1][1] = wmma_bf16(a1, b1, acc[1][1]);
    }
    __syncthreads();
  }

#pragma unroll
  for (int i = 0; i < 2; ++i)
#pragma unroll
    for (int j = 0; j < 2; ++j)
#pragma unroll
      for (int r = 0; r < 8; ++r) {
        int row = wm + 16 * i + r + 8 * hf;
        int col = wn + 16 * j + ln;
        out[(size_t)(mb * 64 + row) * HID_ + nb * 128 + col] = acc[i][j][r];
      }
}

// =====================================================================
extern "C" void kernel_launch(void* const* d_in, const int* in_sizes, int n_in,
                              void* d_out, int out_size, void* d_ws,
                              size_t ws_size, hipStream_t stream) {
  const float* X    = (const float*)d_in[0];  // hidden_states (B,T,HID)
  // d_in[1] = attention_mask (causal -1e9 triangle; applied analytically)
  const float* cosp = (const float*)d_in[2];  // (1,1,T,HD)
  const float* sinp = (const float*)d_in[3];
  const float* Wq   = (const float*)d_in[4];
  const float* Wk   = (const float*)d_in[5];
  const float* Wv   = (const float*)d_in[6];
  const float* Wo   = (const float*)d_in[7];
  float* out        = (float*)d_out;

  // workspace layout (bf16 elements)
  bf16_t* q_ws  = (bf16_t*)d_ws;                                  // B*NH*T*HD
  bf16_t* k_ws  = q_ws + (size_t)B_ * NH_ * T_ * HD_;             // B*KVH*T*HD
  bf16_t* v_ws  = k_ws + (size_t)B_ * KVH_ * T_ * HD_;            // B*KVH*HD*T (transposed)
  bf16_t* ctx_w = v_ws + (size_t)B_ * KVH_ * T_ * HD_;            // B*T*NH*HD

  // 1) QKV projection + RoPE
  dim3 g1((B_ * T_) / 64, 24);
  hipLaunchKernelGGL(k_qkv_rope, g1, dim3(256), 0, stream, X, Wq, Wk, Wv, cosp,
                     sinp, q_ws, k_ws, v_ws);

  // 2) flash attention
  dim3 g2(T_ / 64, NH_, B_);
  hipLaunchKernelGGL(k_attn, g2, dim3(256), 0, stream, q_ws, k_ws, v_ws, ctx_w);

  // 3) output projection
  dim3 g3((B_ * T_) / 64, HID_ / 128);
  hipLaunchKernelGGL(k_outproj, g3, dim3(256), 0, stream, ctx_w, Wo, out);
}
